// CGCNNPyGChargeEarlySelectivePoolingOptimized_74637941670363
// MI455X (gfx1250) — matrix operand order repaired and checked by reading, not compiled, not run on hardware
//
#include <hip/hip_runtime.h>
#include <hip/hip_bf16.h>

using bf16_t = __bf16;
typedef __attribute__((ext_vector_type(16))) __bf16 v16bf;
typedef __attribute__((ext_vector_type(8)))  __bf16 v8bf;
typedef __attribute__((ext_vector_type(8)))  float  v8f;

union V16 { v16bf v; v8bf h[2]; };

__device__ inline float spf(float x) {
    // numerically stable softplus: max(x,0) + log1p(exp(-|x|))
    return fmaxf(x, 0.f) + log1pf(__expf(-fabsf(x)));
}

__device__ inline v8f wmma_bf16(v16bf a, v16bf b, v8f c) {
    return __builtin_amdgcn_wmma_f32_16x16x32_bf16(false, a, false, b, (short)0, c, false, false);
}

// B fragments pre-swizzled: wp[((s*4+t)*32 + lane)*16 + j], lane-contiguous 32B
__device__ inline v16bf load_bfrag(const bf16_t* __restrict__ wp, int s, int t, int lane) {
    const bf16_t* p = wp + (size_t)(((s * 4 + t) * 32 + lane) * 16);
    V16 u;
    u.h[0] = *(const v8bf*)p;
    u.h[1] = *(const v8bf*)(p + 8);
    return u.v;
}

// ---------------------------------------------------------------------------
// Weight prep: fp32 [layers,K,64] -> bf16 WMMA B-fragment layout
// lane L: N = 16*t + (L&15), K = 32*s + 16*(L>>4) + j
// ---------------------------------------------------------------------------
__global__ void prep_w_kernel(const float* __restrict__ W, bf16_t* __restrict__ out,
                              int K, int total) {
    int i = blockIdx.x * blockDim.x + threadIdx.x;
    if (i >= total) return;
    int per_layer = K * 64;
    int layer = i / per_layer;
    int rme = i - layer * per_layer;
    int j = rme & 15;
    int L = (rme >> 4) & 31;
    int t = (rme >> 9) & 3;
    int s = rme >> 11;
    int k = s * 32 + 16 * (L >> 4) + j;
    int n = t * 16 + (L & 15);
    out[i] = (bf16_t)W[(size_t)layer * per_layer + (size_t)k * 64 + n];
}

__global__ void zero_kernel(float* __restrict__ p, int n) {
    int i = blockIdx.x * blockDim.x + threadIdx.x;
    if (i < n) p[i] = 0.f;
}

// ---------------------------------------------------------------------------
// Node embedding: h = concat(x, charge_feat[batch]) @ W_atom + b  (fp32 + bf16)
// 256 threads = 4 nodes x 64 channels
// ---------------------------------------------------------------------------
__global__ __launch_bounds__(256) void embed_nodes_kernel(
    const float* __restrict__ x, const float* __restrict__ charge,
    const int* __restrict__ batch,
    const float* __restrict__ Wc, const float* __restrict__ bc,
    const float* __restrict__ Wa, const float* __restrict__ ba,
    float* __restrict__ h, bf16_t* __restrict__ hbf, int N_) {
    __shared__ float xs[4][92];
    int node0 = blockIdx.x * 4;
    for (int idx = threadIdx.x; idx < 4 * 92; idx += 256) {
        int nn = idx / 92, kk = idx - nn * 92;
        int n = node0 + nn;
        if (n < N_) xs[nn][kk] = x[(size_t)n * 92 + kk];
    }
    __syncthreads();
    int nn = threadIdx.x >> 6, d = threadIdx.x & 63;
    int n = node0 + nn;
    if (n >= N_) return;
    float s = ba[d];
    #pragma unroll 4
    for (int k = 0; k < 92; ++k) s += xs[nn][k] * Wa[(size_t)k * 64 + d];
    float cg = charge[batch[n]];
    #pragma unroll
    for (int c = 0; c < 16; ++c) s += (cg * Wc[c] + bc[c]) * Wa[(size_t)(92 + c) * 64 + d];
    h[(size_t)n * 64 + d] = s;
    hbf[(size_t)n * 64 + d] = (bf16_t)s;
}

// ---------------------------------------------------------------------------
// Edge embedding: ea = bf16(edge_attr @ W_bond + b)
// ---------------------------------------------------------------------------
__global__ __launch_bounds__(256) void embed_edges_kernel(
    const float* __restrict__ eattr,
    const float* __restrict__ Wb, const float* __restrict__ bb,
    bf16_t* __restrict__ ea, int E_) {
    __shared__ float es[4][41];
    int e0 = blockIdx.x * 4;
    for (int idx = threadIdx.x; idx < 4 * 41; idx += 256) {
        int nn = idx / 41, kk = idx - nn * 41;
        int e = e0 + nn;
        if (e < E_) es[nn][kk] = eattr[(size_t)e * 41 + kk];
    }
    __syncthreads();
    int nn = threadIdx.x >> 6, d = threadIdx.x & 63;
    int e = e0 + nn;
    if (e >= E_) return;
    float s = bb[d];
    #pragma unroll 4
    for (int k = 0; k < 41; ++k) s += es[nn][k] * Wb[(size_t)k * 64 + d];
    ea[(size_t)e * 64 + d] = (bf16_t)s;
}

// ---------------------------------------------------------------------------
// Fused per-layer edge kernel (WMMA bf16):
//   t1 = sp([hr|hc|ea] @ euW1 + b1); ea' = t1 @ euW2 + b2
//   t2 = sp([hr|ea'] @ nuW1 + b1');  msg = t2 @ nuW2 + b2'
//   atomicAdd h_new[col] += msg; ea <- ea' (in place)
// 128 threads = 4 waves; each wave owns a tile of 16 edges.
// ---------------------------------------------------------------------------
__global__ __launch_bounds__(128) void edge_layer_kernel(
    const bf16_t* __restrict__ hbf, bf16_t* __restrict__ ea,
    const bf16_t* __restrict__ wpe1, const float* __restrict__ be1,
    const bf16_t* __restrict__ wpe2, const float* __restrict__ be2,
    const bf16_t* __restrict__ wpn1, const float* __restrict__ bn1,
    const bf16_t* __restrict__ wpn2, const float* __restrict__ bn2,
    const int* __restrict__ eidx, float* __restrict__ h_new, int E_) {
    // per-wave LDS staging tile: 16 rows x 64 ch bf16, row padded to 72 (144B stride,
    // 16B aligned, conflict-free over 16 rows)
    __shared__ __align__(16) char smem_raw[4 * 2304];

    const int lane = threadIdx.x & 31;
    const int wv = threadIdx.x >> 5;
    const int tile = blockIdx.x * 4 + wv;
    if (tile * 16 >= E_) return;
    const int m = lane & 15;   // edge row within tile == output column group
    const int hf = lane >> 4;  // K-half select
    const int e = tile * 16 + m;
    const int ec = (e < E_) ? e : (E_ - 1);
    const int row = eidx[ec];
    const int col = eidx[E_ + ec];

    bf16_t* st = (bf16_t*)(smem_raw + wv * 2304);

    // ---- A fragments for GEMM1: [hr | hc | ea], K = 192 (6 steps) ----
    V16 a1[6];
    const bf16_t* srcs[3] = { hbf + (size_t)row * 64, hbf + (size_t)col * 64,
                              ea + (size_t)ec * 64 };
    #pragma unroll
    for (int s = 0; s < 6; ++s) {
        const bf16_t* p = srcs[s >> 1] + (s & 1) * 32 + hf * 16;
        a1[s].h[0] = *(const v8bf*)p;
        a1[s].h[1] = *(const v8bf*)(p + 8);
    }

    v8f acc[4];

    // ---- GEMM1 ----
    #pragma unroll
    for (int t = 0; t < 4; ++t) {
        float b = be1[t * 16 + m];
        #pragma unroll
        for (int j = 0; j < 8; ++j) acc[t][j] = b;
    }
    #pragma unroll
    for (int s = 0; s < 6; ++s)
        #pragma unroll
        for (int t = 0; t < 4; ++t)
            acc[t] = wmma_bf16(a1[s].v, load_bfrag(wpe1, s, t, lane), acc[t]);

    // stage softplus(t1): D layout -> row-major LDS tile
    #pragma unroll
    for (int t = 0; t < 4; ++t)
        #pragma unroll
        for (int j = 0; j < 8; ++j)
            st[(j + 8 * hf) * 72 + t * 16 + m] = (bf16_t)spf(acc[t][j]);
    __threadfence_block();

    // ---- GEMM2: ea' = t1 @ euW2 + b ----
    V16 a2[2];
    #pragma unroll
    for (int s = 0; s < 2; ++s) {
        const bf16_t* p = st + m * 72 + s * 32 + hf * 16;
        a2[s].h[0] = *(const v8bf*)p;
        a2[s].h[1] = *(const v8bf*)(p + 8);
    }
    #pragma unroll
    for (int t = 0; t < 4; ++t) {
        float b = be2[t * 16 + m];
        #pragma unroll
        for (int j = 0; j < 8; ++j) acc[t][j] = b;
    }
    #pragma unroll
    for (int s = 0; s < 2; ++s)
        #pragma unroll
        for (int t = 0; t < 4; ++t)
            acc[t] = wmma_bf16(a2[s].v, load_bfrag(wpe2, s, t, lane), acc[t]);

    // stage ea' (bf16): feeds GEMM3 and the global writeback
    #pragma unroll
    for (int t = 0; t < 4; ++t)
        #pragma unroll
        for (int j = 0; j < 8; ++j)
            st[(j + 8 * hf) * 72 + t * 16 + m] = (bf16_t)acc[t][j];
    __threadfence_block();

    // coalesced in-place ea writeback (2 lanes per edge row, 16B chunks)
    {
        int r = lane >> 1, part = lane & 1;
        if (tile * 16 + r < E_) {
            bf16_t* gd = ea + (size_t)(tile * 16 + r) * 64 + part * 32;
            const bf16_t* gs = st + r * 72 + part * 32;
            #pragma unroll
            for (int q = 0; q < 4; ++q)
                *(v8bf*)(gd + q * 8) = *(const v8bf*)(gs + q * 8);
        }
    }

    // ---- GEMM3: t2 = sp([hr | ea'] @ nuW1 + b), K = 128 ----
    V16 a3[2];
    #pragma unroll
    for (int s = 0; s < 2; ++s) {
        const bf16_t* p = st + m * 72 + s * 32 + hf * 16;
        a3[s].h[0] = *(const v8bf*)p;
        a3[s].h[1] = *(const v8bf*)(p + 8);
    }
    #pragma unroll
    for (int t = 0; t < 4; ++t) {
        float b = bn1[t * 16 + m];
        #pragma unroll
        for (int j = 0; j < 8; ++j) acc[t][j] = b;
    }
    #pragma unroll
    for (int t = 0; t < 4; ++t) {
        acc[t] = wmma_bf16(a1[0].v, load_bfrag(wpn1, 0, t, lane), acc[t]);  // hr reused
        acc[t] = wmma_bf16(a1[1].v, load_bfrag(wpn1, 1, t, lane), acc[t]);
        acc[t] = wmma_bf16(a3[0].v, load_bfrag(wpn1, 2, t, lane), acc[t]);
        acc[t] = wmma_bf16(a3[1].v, load_bfrag(wpn1, 3, t, lane), acc[t]);
    }

    // stage softplus(t2)
    #pragma unroll
    for (int t = 0; t < 4; ++t)
        #pragma unroll
        for (int j = 0; j < 8; ++j)
            st[(j + 8 * hf) * 72 + t * 16 + m] = (bf16_t)spf(acc[t][j]);
    __threadfence_block();

    // ---- GEMM4: msg = t2 @ nuW2 + b ----
    V16 a4[2];
    #pragma unroll
    for (int s = 0; s < 2; ++s) {
        const bf16_t* p = st + m * 72 + s * 32 + hf * 16;
        a4[s].h[0] = *(const v8bf*)p;
        a4[s].h[1] = *(const v8bf*)(p + 8);
    }
    #pragma unroll
    for (int t = 0; t < 4; ++t) {
        float b = bn2[t * 16 + m];
        #pragma unroll
        for (int j = 0; j < 8; ++j) acc[t][j] = b;
    }
    #pragma unroll
    for (int s = 0; s < 2; ++s)
        #pragma unroll
        for (int t = 0; t < 4; ++t)
            acc[t] = wmma_bf16(a4[s].v, load_bfrag(wpn2, s, t, lane), acc[t]);

    // ---- scatter: h_new[col[e]] += msg (fp32 atomics from D layout) ----
    int cols[8];
    #pragma unroll
    for (int j = 0; j < 8; ++j) {
        int ee = tile * 16 + 8 * hf + j;
        cols[j] = (ee < E_) ? eidx[E_ + ee] : -1;
    }
    #pragma unroll
    for (int t = 0; t < 4; ++t) {
        int n = t * 16 + m;
        #pragma unroll
        for (int j = 0; j < 8; ++j)
            if (cols[j] >= 0)
                atomicAdd(&h_new[(size_t)cols[j] * 64 + n], acc[t][j]);
    }
}

// ---------------------------------------------------------------------------
// BatchNorm: per-channel sum/sumsq reduction, then apply + softplus + residual
// ---------------------------------------------------------------------------
__global__ __launch_bounds__(256) void bn_reduce_kernel(const float* __restrict__ hn,
                                                        float* __restrict__ acc, int N_) {
    int c = threadIdx.x & 63;
    int sub = threadIdx.x >> 6;  // 0..3
    float s = 0.f, ss = 0.f;
    for (int n = blockIdx.x * 4 + sub; n < N_; n += gridDim.x * 4) {
        float v = hn[(size_t)n * 64 + c];
        s += v; ss += v * v;
    }
    __shared__ float red[2][4][64];
    red[0][sub][c] = s; red[1][sub][c] = ss;
    __syncthreads();
    if (sub == 0) {
        s  = red[0][0][c] + red[0][1][c] + red[0][2][c] + red[0][3][c];
        ss = red[1][0][c] + red[1][1][c] + red[1][2][c] + red[1][3][c];
        atomicAdd(&acc[c], s);
        atomicAdd(&acc[64 + c], ss);
    }
}

__global__ void bn_apply_kernel(const float* __restrict__ hn, float* __restrict__ h,
                                bf16_t* __restrict__ hbf, const float* __restrict__ acc,
                                const float* __restrict__ gamma, const float* __restrict__ beta,
                                int N_) {
    int i = blockIdx.x * blockDim.x + threadIdx.x;
    if (i >= N_ * 64) return;
    int c = i & 63;
    float invN = 1.f / (float)N_;
    float mu = acc[c] * invN;
    float var = acc[64 + c] * invN - mu * mu;
    float v = (hn[i] - mu) * rsqrtf(var + 1e-5f) * gamma[c] + beta[c];
    float r = spf(v) + h[i];
    h[i] = r;
    hbf[i] = (bf16_t)r;
}

// ---------------------------------------------------------------------------
// Global mean pool (atomics) + MLP predictor (block per graph)
// ---------------------------------------------------------------------------
__global__ void pool_kernel(const float* __restrict__ h, const int* __restrict__ batch,
                            float* __restrict__ sums, float* __restrict__ cnts, int N_) {
    int i = blockIdx.x * blockDim.x + threadIdx.x;
    if (i >= N_ * 64) return;
    int n = i >> 6, c = i & 63;
    int g = batch[n];
    atomicAdd(&sums[(size_t)g * 64 + c], h[i]);
    if (c == 0) atomicAdd(&cnts[g], 1.f);
}

__global__ __launch_bounds__(128) void predict_kernel(
    const float* __restrict__ sums, const float* __restrict__ cnts,
    const float* __restrict__ W1, const float* __restrict__ b1,
    const float* __restrict__ W2, const float* __restrict__ b2,
    const float* __restrict__ W3, const float* __restrict__ b3,
    float* __restrict__ out) {
    __shared__ float gr[64], z1[128], z2[128];
    int g = blockIdx.x, t = threadIdx.x;
    if (t < 64) gr[t] = sums[(size_t)g * 64 + t] / fmaxf(cnts[g], 1.f);
    __syncthreads();
    float s = b1[t];
    #pragma unroll 4
    for (int d = 0; d < 64; ++d) s += gr[d] * W1[(size_t)d * 128 + t];
    z1[t] = spf(s);
    __syncthreads();
    float s2 = b2[t];
    #pragma unroll 4
    for (int k = 0; k < 128; ++k) s2 += z1[k] * W2[(size_t)k * 128 + t];
    z2[t] = spf(s2);
    __syncthreads();
    if (t == 0) {
        float o = b3[0];
        for (int k = 0; k < 128; ++k) o += z2[k] * W3[k];
        out[g] = o;
    }
}

// ---------------------------------------------------------------------------
extern "C" void kernel_launch(void* const* d_in, const int* in_sizes, int n_in,
                              void* d_out, int out_size, void* d_ws, size_t ws_size,
                              hipStream_t stream) {
    const float* x      = (const float*)d_in[0];
    const float* eattr  = (const float*)d_in[1];
    const float* charge = (const float*)d_in[2];
    const int*   eidx   = (const int*)d_in[3];
    const int*   batch  = (const int*)d_in[4];
    const float* Wc  = (const float*)d_in[5];
    const float* bc  = (const float*)d_in[6];
    const float* Wa  = (const float*)d_in[7];
    const float* ba  = (const float*)d_in[8];
    const float* Wb  = (const float*)d_in[9];
    const float* bb  = (const float*)d_in[10];
    const float* nuW1 = (const float*)d_in[11];
    const float* nub1 = (const float*)d_in[12];
    const float* nuW2 = (const float*)d_in[13];
    const float* nub2 = (const float*)d_in[14];
    const float* euW1 = (const float*)d_in[15];
    const float* eub1 = (const float*)d_in[16];
    const float* euW2 = (const float*)d_in[17];
    const float* eub2 = (const float*)d_in[18];
    const float* gam  = (const float*)d_in[19];
    const float* bet  = (const float*)d_in[20];
    const float* pW1  = (const float*)d_in[21];
    const float* pb1  = (const float*)d_in[22];
    const float* pW2  = (const float*)d_in[23];
    const float* pb2  = (const float*)d_in[24];
    const float* pW3  = (const float*)d_in[25];
    const float* pb3  = (const float*)d_in[26];
    float* out = (float*)d_out;

    const int N = in_sizes[0] / 92;
    const int E = in_sizes[1] / 41;
    const int G = in_sizes[2];

    char* ws = (char*)d_ws;
    size_t off = 0;
    auto carve = [&](size_t bytes) -> char* {
        char* p = ws + off;
        off += (bytes + 255) & ~(size_t)255;
        return p;
    };
    float*  h     = (float*)carve((size_t)N * 64 * 4);
    float*  h_new = (float*)carve((size_t)N * 64 * 4);
    bf16_t* hbf   = (bf16_t*)carve((size_t)N * 64 * 2);
    bf16_t* ea    = (bf16_t*)carve((size_t)E * 64 * 2);
    bf16_t* wpe1  = (bf16_t*)carve((size_t)3 * 192 * 64 * 2);
    bf16_t* wpe2  = (bf16_t*)carve((size_t)3 * 64 * 64 * 2);
    bf16_t* wpn1  = (bf16_t*)carve((size_t)3 * 128 * 64 * 2);
    bf16_t* wpn2  = (bf16_t*)carve((size_t)3 * 64 * 64 * 2);
    float*  bnacc = (float*)carve(128 * 4);
    float*  psum  = (float*)carve((size_t)G * 64 * 4);
    float*  pcnt  = (float*)carve((size_t)G * 4);

    // weight fragment prep (tiny)
    prep_w_kernel<<<(3 * 192 * 64 + 255) / 256, 256, 0, stream>>>(euW1, wpe1, 192, 3 * 192 * 64);
    prep_w_kernel<<<(3 * 64 * 64 + 255) / 256, 256, 0, stream>>>(euW2, wpe2, 64, 3 * 64 * 64);
    prep_w_kernel<<<(3 * 128 * 64 + 255) / 256, 256, 0, stream>>>(nuW1, wpn1, 128, 3 * 128 * 64);
    prep_w_kernel<<<(3 * 64 * 64 + 255) / 256, 256, 0, stream>>>(nuW2, wpn2, 64, 3 * 64 * 64);

    // embeddings
    embed_nodes_kernel<<<(N + 3) / 4, 256, 0, stream>>>(x, charge, batch, Wc, bc, Wa, ba, h, hbf, N);
    embed_edges_kernel<<<(E + 3) / 4, 256, 0, stream>>>(eattr, Wb, bb, ea, E);

    const int tiles = (E + 15) / 16;
    const int eblocks = (tiles + 3) / 4;

    for (int i = 0; i < 3; ++i) {
        zero_kernel<<<(N * 64 + 255) / 256, 256, 0, stream>>>(h_new, N * 64);
        zero_kernel<<<1, 128, 0, stream>>>(bnacc, 128);
        edge_layer_kernel<<<eblocks, 128, 0, stream>>>(
            hbf, ea,
            wpe1 + (size_t)i * 192 * 64, eub1 + i * 64,
            wpe2 + (size_t)i * 64 * 64,  eub2 + i * 64,
            wpn1 + (size_t)i * 128 * 64, nub1 + i * 64,
            wpn2 + (size_t)i * 64 * 64,  nub2 + i * 64,
            eidx, h_new, E);
        bn_reduce_kernel<<<256, 256, 0, stream>>>(h_new, bnacc, N);
        bn_apply_kernel<<<(N * 64 + 255) / 256, 256, 0, stream>>>(
            h_new, h, hbf, bnacc, gam + i * 64, bet + i * 64, N);
    }

    zero_kernel<<<(G * 64 + 255) / 256, 256, 0, stream>>>(psum, G * 64);
    zero_kernel<<<1, 256, 0, stream>>>(pcnt, G);
    pool_kernel<<<(N * 64 + 255) / 256, 256, 0, stream>>>(h, batch, psum, pcnt, N);
    predict_kernel<<<G, 128, 0, stream>>>(psum, pcnt, pW1, pb1, pW2, pb2, pW3, pb3, out);
}
